// DemoModule_25512105739100
// MI455X (gfx1250) — matrix operand order repaired and checked by reading, not compiled, not run on hardware
//
#include <hip/hip_runtime.h>
#include <hip/hip_bf16.h>
#include <stdint.h>

typedef __attribute__((ext_vector_type(16))) _Float16 v16h;
typedef __attribute__((ext_vector_type(8)))  _Float16 v8h;
typedef __attribute__((ext_vector_type(8)))  float    v8f;

#define BATCH   16384
#define FEAT    64
#define BAG     10
#define EDIM    16
#define DWIDTH  1024
#define EPSV    1e-5f

// ---------------------------------------------------------------- utilities
__global__ void zero_f32(float* __restrict__ p, int n) {
    int i = blockIdx.x * blockDim.x + threadIdx.x;
    if (i < n) p[i] = 0.f;
}

// W[K][N] (row-major, as used in x @ W) -> WT[N][K] in f16, write-coalesced.
__global__ void transpose_f32_to_f16(const float* __restrict__ W,
                                     _Float16* __restrict__ WT,
                                     int K, int N) {
    int gid = blockIdx.x * blockDim.x + threadIdx.x;
    if (gid >= K * N) return;
    int n = gid / K;
    int k = gid - n * K;
    WT[(size_t)n * K + k] = (_Float16)W[(size_t)k * N + n];
}

// ------------------------------------------------- embedding gather + pool
// One thread per output element x[b, d]; 16 lanes of a half-wave share (b,f)
// and fetch one contiguous 64B embedding row per bag slot (table is 64 MB,
// L2-resident on the 192 MB L2).
__global__ void embed_pool(const int* __restrict__ idx,
                           const float* __restrict__ emb,
                           float* __restrict__ x) {
    int gid = blockIdx.x * blockDim.x + threadIdx.x;   // over B*D, exact grid
    int b = gid >> 10;
    int d = gid & (DWIDTH - 1);
    int f = d >> 4;
    int e = d & (EDIM - 1);
    const int* ip = idx + ((size_t)b * FEAT + f) * BAG;
    float s = 0.f;
#pragma unroll
    for (int l = 0; l < BAG; ++l) {
        int v = ip[l];
        s += emb[(size_t)v * EDIM + e];
    }
    x[gid] = s;
}

// ------------------------------------------------------- batchnorm stats
__global__ void col_stats(const float* __restrict__ x,
                          float* __restrict__ sum, float* __restrict__ sumsq) {
    int r0 = blockIdx.x * 64;
    int t  = threadIdx.x;
    float s[4] = {0.f, 0.f, 0.f, 0.f};
    float q[4] = {0.f, 0.f, 0.f, 0.f};
    for (int r = 0; r < 64; ++r) {
        const float* row = x + (size_t)(r0 + r) * DWIDTH;
#pragma unroll
        for (int c = 0; c < 4; ++c) {
            float v = row[t + c * 256];
            s[c] += v;
            q[c] += v * v;
        }
    }
#pragma unroll
    for (int c = 0; c < 4; ++c) {
        atomicAdd(&sum[t + c * 256],   s[c]);
        atomicAdd(&sumsq[t + c * 256], q[c]);
    }
}

__global__ void finalize_stats(const float* __restrict__ sum,
                               const float* __restrict__ sumsq,
                               float* __restrict__ mean, float* __restrict__ rstd) {
    int i = blockIdx.x * blockDim.x + threadIdx.x;
    if (i >= DWIDTH) return;
    float m = sum[i] * (1.f / BATCH);
    float v = sumsq[i] * (1.f / BATCH) - m * m;   // population variance (ddof=0)
    mean[i] = m;
    rstd[i] = rsqrtf(v + EPSV);
}

__global__ void normalize_f16(const float* __restrict__ x,
                              const float* __restrict__ mean,
                              const float* __restrict__ rstd,
                              _Float16* __restrict__ xh) {
    int gid = blockIdx.x * blockDim.x + threadIdx.x;
    int d = gid & (DWIDTH - 1);
    xh[gid] = (_Float16)((x[gid] - mean[d]) * rstd[d]);
}

// ------------------------------------------------------------ WMMA GEMM v3
__device__ __forceinline__ v16h concat8(v8h lo, v8h hi) {
    return __builtin_shufflevector(lo, hi,
        0, 1, 2, 3, 4, 5, 6, 7, 8, 9, 10, 11, 12, 13, 14, 15);
}

// gfx1250 async copy: 16B global -> LDS, tracked by ASYNCcnt (ISA 10.x / 15.18.3).
// VDST operand carries the LDS byte offset; address pair is the global address.
__device__ __forceinline__ void async_cp16(uint32_t lds_off, const _Float16* g) {
    asm volatile("global_load_async_to_lds_b128 %0, %1, off"
                 :: "v"(lds_off), "v"(g) : "memory");
}
__device__ __forceinline__ void wait_async0() {
    asm volatile("s_wait_asynccnt 0x0" ::: "memory");
}

#define KSTEP   32
#define ROW_H   40                    // halves per LDS row: 32 + 8 pad (80 B)
#define TILE_H  (128 * ROW_H)         // one 128x32 f16 tile in LDS (10240 B)

// C[M,N] = relu(A[M,K] @ WT[N,K]^T + bias); A/WT f16, f32 accumulate, C f16.
// Block: 256 threads (8 waves), tile M=128 x N=128, K-step 32.
// LDS double-buffered A/B tiles filled by async-to-LDS; each wave computes a
// 32x64 strip = 2(M) x 4(N) WMMA accumulators.  All 12 fragment ds_loads are
// issued before the 8 back-to-back v_wmma per K-step so dscnt waits stay
// partial and the XDL pipe is kept busy.
template <int N, int K>
__global__ void __launch_bounds__(256, 2)
gemm_bias_relu_wmma(const _Float16* __restrict__ A,
                    const _Float16* __restrict__ WT,
                    const float* __restrict__ bias,
                    _Float16* __restrict__ C) {
    __shared__ _Float16 smem[4 * TILE_H];   // [A0 | A1 | B0 | B1] = 40 KB

    const int lane = threadIdx.x & 31;
    const int wave = threadIdx.x >> 5;
    const int l15  = lane & 15;
    const int hi   = lane >> 4;
    const int m_base = blockIdx.x * 128 + (wave & 3) * 32;   // wave's 32 rows
    const int n_base = blockIdx.y * 128 + (wave >> 2) * 64;  // wave's 64 cols

    const _Float16* gA = A  + (size_t)(blockIdx.x * 128) * K;  // 128 rows
    const _Float16* gB = WT + (size_t)(blockIdx.y * 128) * K;  // 128 cols of W

    const uint32_t ldsA0 = (uint32_t)(uintptr_t)&smem[0];
    const uint32_t ldsB0 = (uint32_t)(uintptr_t)&smem[2 * TILE_H];

    // stage tile for K-slice kk into buffer buf (4 async b128 per thread)
    auto stage = [&](int kk, int buf) {
        const uint32_t dA = ldsA0 + (uint32_t)buf * (TILE_H * 2);
        const uint32_t dB = ldsB0 + (uint32_t)buf * (TILE_H * 2);
#pragma unroll
        for (int h = 0; h < 2; ++h) {
            int c   = threadIdx.x + h * 256;    // chunk id 0..511
            int row = c >> 2;                   // 0..127
            int kc  = c & 3;                    // 16B chunk within K-slice
            uint32_t lo = (uint32_t)(row * 80 + kc * 16);
            async_cp16(dA + lo, gA + (size_t)row * K + kk + kc * 8);
            async_cp16(dB + lo, gB + (size_t)row * K + kk + kc * 8);
        }
    };

    v8f acc[2][4] = {};

    stage(0, 0);
    wait_async0();
    __syncthreads();

    for (int kk = 0; kk < K; kk += KSTEP) {
        const int buf = (kk / KSTEP) & 1;
        if (kk + KSTEP < K) stage(kk + KSTEP, buf ^ 1);   // overlap with math

        const _Float16* sA = smem + buf * TILE_H;
        const _Float16* sB = smem + (2 + buf) * TILE_H;

        // ---- issue ALL fragment loads first (one ds clause) ----
        // A fragments: lane holds row M=lane&15; chunks at k=(hi?8:0), +16
        v16h a[2];
#pragma unroll
        for (int mi = 0; mi < 2; ++mi) {
            const _Float16* p = sA + (size_t)((wave & 3) * 32 + mi * 16 + l15) * ROW_H + hi * 8;
            a[mi] = concat8(*(const v8h*)p, *(const v8h*)(p + 16));
        }
        // B fragments: lane holds col N=lane&15; chunks at k=(hi?8:0), +16
        v16h b[4];
#pragma unroll
        for (int t = 0; t < 4; ++t) {
            const _Float16* p = sB + (size_t)((wave >> 2) * 64 + t * 16 + l15) * ROW_H + hi * 8;
            b[t] = concat8(*(const v8h*)p, *(const v8h*)(p + 16));
        }

        // ---- 8 back-to-back WMMAs ----
#pragma unroll
        for (int t = 0; t < 4; ++t)
#pragma unroll
            for (int mi = 0; mi < 2; ++mi)
                acc[mi][t] = __builtin_amdgcn_wmma_f32_16x16x32_f16(
                                 false, a[mi], false, b[t], (short)0, acc[mi][t],
                                 false, false);

        if (kk + KSTEP < K) {
            wait_async0();      // next tiles landed in LDS
            __syncthreads();    // everyone done reading current buffers
        }
    }

    // epilogue: C/D layout -> VGPR r: M = r + 8*(lane>=16), N = lane&15
#pragma unroll
    for (int mi = 0; mi < 2; ++mi) {
        const int rg0 = m_base + mi * 16 + hi * 8;
#pragma unroll
        for (int t = 0; t < 4; ++t) {
            const int col = n_base + t * 16 + l15;
            const float bv = bias[col];
#pragma unroll
            for (int r = 0; r < 8; ++r) {
                float v = acc[mi][t][r] + bv;
                v = v > 0.f ? v : 0.f;
                C[(size_t)(rg0 + r) * N + col] = (_Float16)v;
            }
        }
    }
}

// ----------------------------------------------- final GEMV (512 -> 1) + sigmoid
__global__ void gemv_sigmoid(const _Float16* __restrict__ H,
                             const float* __restrict__ W3,
                             const float* __restrict__ b3,
                             float* __restrict__ out) {
    const int lane = threadIdx.x & 31;
    const int wave = threadIdx.x >> 5;
    const int row  = blockIdx.x * 8 + wave;
    const _Float16* h = H + (size_t)row * 512;
    float s = 0.f;
#pragma unroll
    for (int k = 0; k < 512; k += 32)
        s = fmaf((float)h[k + lane], W3[k + lane], s);
#pragma unroll
    for (int off = 16; off >= 1; off >>= 1)
        s += __shfl_xor(s, off, 32);
    if (lane == 0)
        out[row] = 1.f / (1.f + expf(-(s + b3[0])));
}

// ----------------------------------------------------------------- launcher
extern "C" void kernel_launch(void* const* d_in, const int* in_sizes, int n_in,
                              void* d_out, int out_size, void* d_ws, size_t ws_size,
                              hipStream_t stream) {
    const int*   idx = (const int*)  d_in[0];
    const float* emb = (const float*)d_in[1];
    const float* W1  = (const float*)d_in[2];
    const float* b1  = (const float*)d_in[3];
    const float* W2  = (const float*)d_in[4];
    const float* b2  = (const float*)d_in[5];
    const float* W3  = (const float*)d_in[6];
    const float* b3  = (const float*)d_in[7];
    float* out = (float*)d_out;

    // workspace layout (all 256B aligned)
    char* ws = (char*)d_ws;
    float*    sum   = (float*)(ws + 0);                         //  4 KB
    float*    sumsq = (float*)(ws + 4096);                      //  4 KB
    float*    meanp = (float*)(ws + 8192);                      //  4 KB
    float*    rstdp = (float*)(ws + 12288);                     //  4 KB
    _Float16* W1T   = (_Float16*)(ws + 16384);                  //  2 MiB [1024][1024]
    _Float16* W2T   = (_Float16*)(ws + 16384 + (size_t)2097152);//  1 MiB [512][1024]
    float*    x     = (float*)(ws + 16384 + (size_t)3145728);   // 64 MiB [B][1024] f32
    _Float16* xh    = (_Float16*)((char*)x + (size_t)BATCH * DWIDTH * 4); // 32 MiB
    // x is dead after normalize: reuse its region for h1 (32 MiB) + h2 (16 MiB)
    _Float16* h1    = (_Float16*)x;
    _Float16* h2    = (_Float16*)((char*)x + (size_t)BATCH * DWIDTH * 2);

    zero_f32<<<8, 256, 0, stream>>>(sum, 2048);  // zeros sum + sumsq (contiguous)

    transpose_f32_to_f16<<<(1024 * 1024) / 256, 256, 0, stream>>>(W1, W1T, 1024, 1024);
    transpose_f32_to_f16<<<(512  * 1024) / 256, 256, 0, stream>>>(W2, W2T, 1024, 512);

    embed_pool<<<(BATCH * DWIDTH) / 256, 256, 0, stream>>>(idx, emb, x);

    col_stats<<<BATCH / 64, 256, 0, stream>>>(x, sum, sumsq);
    finalize_stats<<<4, 256, 0, stream>>>(sum, sumsq, meanp, rstdp);
    normalize_f16<<<(BATCH * DWIDTH) / 256, 256, 0, stream>>>(x, meanp, rstdp, xh);

    gemm_bias_relu_wmma<1024, 1024>
        <<<dim3(BATCH / 128, 1024 / 128), 256, 0, stream>>>(xh, W1T, b1, h1);
    gemm_bias_relu_wmma<512, 1024>
        <<<dim3(BATCH / 128, 512 / 128), 256, 0, stream>>>(h1, W2T, b2, h2);

    gemv_sigmoid<<<BATCH / 8, 256, 0, stream>>>(h2, W3, b3, out);
}
